// KroneckerLSTM_66477503807912
// MI455X (gfx1250) — compile-verified, matrix operand order checked by reading, not run on hardware
//
#include <hip/hip_runtime.h>
#include <hip/hip_bf16.h>

// ---------------------------------------------------------------------------
// KroneckerLSTM on MI455X (gfx1250, wave32, WMMA bf16 + TDM tile DMA):
//   pass A: convert L_g -> bf16 row-major; convert+transpose x,h,R_g -> bf16
//   pass B: U_g  = L_g @ X          (bf16 x bf16 -> f32 acc -> bf16)
//   pass C: pre_p = U_x@R_x + U_h@R_h + b_x + b_h        (f32 out)
//   pass D: elementwise LSTM cell -> h_new, c_new
// GEMM: 128x128 block tile, 8 waves, 64x32 wave tile (4x2 wmma accum), BK=32,
// ping-pong LDS. Tile fills via TENSOR_LOAD_TO_LDS (TDM) with LDS padding
// (16 DW data + 8 DW pad = 48 bf16 row stride); TENSORcnt double buffering.
// ---------------------------------------------------------------------------

typedef __attribute__((ext_vector_type(16))) __bf16 bf16x16;
typedef __attribute__((ext_vector_type(8)))  float  f32x8;
typedef __attribute__((ext_vector_type(4)))  unsigned int u32x4;
typedef __attribute__((ext_vector_type(8)))  int    i32x8;
typedef __attribute__((ext_vector_type(4)))  int    i32x4;

#define TDIM 1024
#define NM   ((size_t)TDIM * TDIM)
#define BM 128
#define BN 128
#define BK 32
#define LDS_STRIDE 48   // 48 bf16 = 96 B rows (16 DW data + 8 DW pad)
#define NTHREADS 256

#if __has_builtin(__builtin_amdgcn_tensor_load_to_lds) && \
    __has_builtin(__builtin_amdgcn_s_wait_tensorcnt)
#define KLSTM_TDM 1
#else
#define KLSTM_TDM 0
#endif

struct Ptrs8  { const float* p[8];  };
struct Ptrs10 { const float* p[10]; };

// --- pass A1: f32 -> bf16, row-major copy (L matrices) ---------------------
__global__ __launch_bounds__(NTHREADS) void klstm_conv(Ptrs8 a,
                                                       __bf16* __restrict__ out) {
  const int z = blockIdx.z;
  const float* __restrict__ src = a.p[z];
  __bf16* __restrict__ dst = out + (size_t)z * NM;
  const size_t base = ((size_t)blockIdx.x * NTHREADS + threadIdx.x) * 8;
  const float4 v0 = *(const float4*)(src + base);
  const float4 v1 = *(const float4*)(src + base + 4);
  union { __bf16 h[8]; uint4 q; } u;
  u.h[0] = (__bf16)v0.x; u.h[1] = (__bf16)v0.y;
  u.h[2] = (__bf16)v0.z; u.h[3] = (__bf16)v0.w;
  u.h[4] = (__bf16)v1.x; u.h[5] = (__bf16)v1.y;
  u.h[6] = (__bf16)v1.z; u.h[7] = (__bf16)v1.w;
  *(uint4*)(dst + base) = u.q;
}

// --- pass A2: f32 -> bf16 with transpose (x, h, R matrices) ----------------
// out[n*TDIM + k] = (bf16) src[k*TDIM + n]
__global__ __launch_bounds__(NTHREADS) void klstm_convT(Ptrs10 a,
                                                        __bf16* __restrict__ out) {
  __shared__ float t[64][65];
  const int z = blockIdx.z;
  const float* __restrict__ src = a.p[z];
  __bf16* __restrict__ dst = out + (size_t)z * NM;
  const int r0 = blockIdx.y * 64, c0 = blockIdx.x * 64;
  const int tid = threadIdx.x;
#pragma unroll
  for (int j = 0; j < 4; ++j) {
    int c  = tid + j * NTHREADS;
    int rr = c >> 4;
    int cc = (c & 15) * 4;
    const float4 v = *(const float4*)(src + (size_t)(r0 + rr) * TDIM + c0 + cc);
    t[rr][cc] = v.x; t[rr][cc + 1] = v.y;
    t[rr][cc + 2] = v.z; t[rr][cc + 3] = v.w;
  }
  __syncthreads();
#pragma unroll
  for (int j = 0; j < 2; ++j) {
    int c  = tid + j * NTHREADS;
    int i  = c >> 3;        // output row  (= src col)
    int jj = (c & 7) * 8;   // output col  (= src row)
    union { __bf16 h[8]; uint4 q; } u;
#pragma unroll
    for (int e = 0; e < 8; ++e) u.h[e] = (__bf16)t[jj + e][i];
    *(uint4*)(dst + (size_t)(c0 + i) * TDIM + r0 + jj) = u.q;
  }
}

// --- WMMA fragment helpers -------------------------------------------------
__device__ inline bf16x16 load_frag(const __bf16* base) {
  union { bf16x16 v; uint4 q[2]; } u;
  u.q[0] = *(const uint4*)(base);
  u.q[1] = *(const uint4*)(base + 16);
  return u.v;
}

__device__ inline f32x8 wmma_bf16(bf16x16 a, bf16x16 b, f32x8 c) {
  return __builtin_amdgcn_wmma_f32_16x16x32_bf16(
      false, a, false, b, (short)0, c, false, false);
}

__device__ inline void compute_tiles(const __bf16* lAb, const __bf16* lBb,
                                     f32x8 acc[4][2], int wm, int wn,
                                     int lrow, int kb) {
  bf16x16 af[4], bfr[2];
#pragma unroll
  for (int i = 0; i < 4; ++i)
    af[i] = load_frag(lAb + (wm * 64 + i * 16 + lrow) * LDS_STRIDE + kb);
#pragma unroll
  for (int jj = 0; jj < 2; ++jj)
    bfr[jj] = load_frag(lBb + (wn * 32 + jj * 16 + lrow) * LDS_STRIDE + kb);
#pragma unroll
  for (int i = 0; i < 4; ++i)
#pragma unroll
    for (int jj = 0; jj < 2; ++jj)
      acc[i][jj] = wmma_bf16(af[i], bfr[jj], acc[i][jj]);
}

#if KLSTM_TDM
// --- TDM descriptor: 2D tile, 128 rows x 32 bf16, into padded LDS ----------
// D# group0: count=1 | lds_addr | global_addr | type=2
__device__ inline u32x4 tdm_g0(unsigned lds_off, const __bf16* gaddr) {
  const unsigned long long ga = (unsigned long long)(size_t)gaddr;
  u32x4 g;
  g.x = 1u;                                       // count=1 (valid descriptor)
  g.y = lds_off;                                  // lds_addr (bytes)
  g.z = (unsigned)ga;                             // global_addr[31:0]
  g.w = ((unsigned)(ga >> 32) & 0x01FFFFFFu)      // global_addr[56:32]
        | (2u << 30);                             // type=2 ("image")
  return g;
}
// D# group1: data_size=2B, pad 8 DW every 16 DW; tensor 1024x1024,
// tile 32(x) x 128(y), dim0 stride 1024 elements.
__device__ inline i32x8 tdm_g1() {
  i32x8 g;
  g[0] = (1 << 16)      // data_size = 1 -> 2 bytes
       | (1 << 20)      // pad_enable
       | (3 << 22)      // pad_interval: code 3 -> 16 DWORDs (= 32 bf16 row)
       | (7 << 25);     // pad_amount:   code 7 ->  8 DWORDs (= 16 bf16 pad)
  g[1] = (TDIM & 0xFFFF) << 16;                          // tensor_dim0 lo16
  g[2] = ((TDIM >> 16) & 0xFFFF) | ((TDIM & 0xFFFF) << 16); // dim0 hi|dim1 lo
  g[3] = ((TDIM >> 16) & 0xFFFF) | (BK << 16);           // dim1 hi | tile_dim0
  g[4] = BM;                                             // tile_dim1 (tile_dim2=0)
  g[5] = TDIM;                                           // tensor_dim0_stride lo32
  g[6] = 0;
  g[7] = 0;
  return g;
}

__device__ inline void tdm_fill(const __bf16* src, int row0, int k0,
                                unsigned lds_off) {
  i32x4 z4 = 0;
  i32x8 z8 = 0;
  // 6-arg form: (g0 u32x4, g1 i32x8, g2 i32x4, g3 i32x4, i32x8, cpol)
  __builtin_amdgcn_tensor_load_to_lds(
      tdm_g0(lds_off, src + (size_t)row0 * TDIM + k0), tdm_g1(), z4, z4, z8, 0);
}
#endif  // KLSTM_TDM

#if !KLSTM_TDM
// --- fallback: register-staged tile copy (struct by value, no arrays) ------
struct TileRegs { uint4 q0, q1; };

__device__ inline TileRegs load_tile(const __bf16* __restrict__ src, int row0,
                                     int k0, int tid) {
  TileRegs t;
  const int c0 = tid, c1 = tid + NTHREADS;
  t.q0 = *(const uint4*)(src + (size_t)(row0 + (c0 >> 2)) * TDIM + k0 + (c0 & 3) * 8);
  t.q1 = *(const uint4*)(src + (size_t)(row0 + (c1 >> 2)) * TDIM + k0 + (c1 & 3) * 8);
  return t;
}
__device__ inline void store_tile(__bf16* dst, int tid, TileRegs t) {
  const int c0 = tid, c1 = tid + NTHREADS;
  *(uint4*)(dst + (c0 >> 2) * LDS_STRIDE + (c0 & 3) * 8) = t.q0;
  *(uint4*)(dst + (c1 >> 2) * LDS_STRIDE + (c1 & 3) * 8) = t.q1;
}
#endif

// one full K pass: acc += A[row0:+128, :] * Bt[col0:+128, :]^T
__device__ inline void gemm_pass(const __bf16* __restrict__ A,
                                 const __bf16* __restrict__ Bt,
                                 int row0, int col0,
                                 __bf16* lA0, __bf16* lA1,
                                 __bf16* lB0, __bf16* lB1,
                                 f32x8 acc[4][2], int tid) {
  const int lane = tid & 31;
  const int wv   = tid >> 5;
  const int wm   = wv >> 2;
  const int wn   = wv & 3;
  const int lrow = lane & 15;
  const int kb   = (lane >> 4) * 8;

#if KLSTM_TDM
  // wave-uniform SGPR wave id -> scalar branch (TDM ignores EXEC, must branch)
  const int wid = __builtin_amdgcn_readfirstlane((unsigned)tid) >> 5;
  const unsigned a0 = (unsigned)(size_t)lA0, a1 = (unsigned)(size_t)lA1;
  const unsigned b0 = (unsigned)(size_t)lB0, b1 = (unsigned)(size_t)lB1;

  if (wid == 0) {
    tdm_fill(A, row0, 0, a0);   tdm_fill(Bt, col0, 0, b0);
    tdm_fill(A, row0, BK, a1);  tdm_fill(Bt, col0, BK, b1);
    __builtin_amdgcn_s_wait_tensorcnt(2);   // buf0 pair complete
  }
  __syncthreads();

  for (int kt = 0; kt < TDIM; kt += 2 * BK) {
    compute_tiles(lA0, lB0, acc, wm, wn, lrow, kb);     // tile kt
    __syncthreads();                                    // all done reading buf0
    if (wid == 0) {
      const int k2 = kt + 2 * BK;
      if (k2 < TDIM) {
        tdm_fill(A, row0, k2, a0);  tdm_fill(Bt, col0, k2, b0);
        __builtin_amdgcn_s_wait_tensorcnt(2);           // buf1 pair complete
      } else {
        __builtin_amdgcn_s_wait_tensorcnt(0);
      }
    }
    __syncthreads();                                    // buf1 published
    compute_tiles(lA1, lB1, acc, wm, wn, lrow, kb);     // tile kt+BK
    __syncthreads();
    if (wid == 0) {
      const int k3 = kt + 3 * BK;
      if (k3 < TDIM) {
        tdm_fill(A, row0, k3, a1);  tdm_fill(Bt, col0, k3, b1);
        __builtin_amdgcn_s_wait_tensorcnt(2);           // buf0 pair complete
      } else {
        __builtin_amdgcn_s_wait_tensorcnt(0);
      }
    }
    __syncthreads();
  }
#else
  TileRegs ra = load_tile(A, row0, 0, tid), rb = load_tile(Bt, col0, 0, tid);
  store_tile(lA0, tid, ra);  store_tile(lB0, tid, rb);
  ra = load_tile(A, row0, BK, tid);  rb = load_tile(Bt, col0, BK, tid);
  __syncthreads();

  for (int kt = 0; kt < TDIM; kt += 2 * BK) {
    store_tile(lA1, tid, ra);  store_tile(lB1, tid, rb);   // tile kt+BK
    const int k2 = (kt + 2 * BK) & (TDIM - 1);
    ra = load_tile(A, row0, k2, tid);  rb = load_tile(Bt, col0, k2, tid);
    compute_tiles(lA0, lB0, acc, wm, wn, lrow, kb);        // tile kt
    __syncthreads();
    store_tile(lA0, tid, ra);  store_tile(lB0, tid, rb);   // tile kt+2BK
    const int k3 = (kt + 3 * BK) & (TDIM - 1);
    ra = load_tile(A, row0, k3, tid);  rb = load_tile(Bt, col0, k3, tid);
    compute_tiles(lA1, lB1, acc, wm, wn, lrow, kb);        // tile kt+BK
    __syncthreads();
  }
#endif
}

// --- pass B: U_g = L_g @ X, bf16 out ---------------------------------------
__global__ __launch_bounds__(NTHREADS) void klstm_gemm1(
    const __bf16* __restrict__ Lbf, const __bf16* __restrict__ Tbf,
    __bf16* __restrict__ U) {
  __shared__ __bf16 lA[2][BM * LDS_STRIDE];
  __shared__ __bf16 lB[2][BN * LDS_STRIDE];

  const int tid  = threadIdx.x;
  const int lane = tid & 31;
  const int g    = blockIdx.z;
  const int row0 = blockIdx.y * BM;
  const int col0 = blockIdx.x * BN;

  const __bf16* A  = Lbf + (size_t)g * NM;
  const __bf16* Bt = Tbf + (size_t)(g & 1) * NM;   // x^T (even g) / h^T (odd)
  __bf16* out      = U + (size_t)g * NM;

  f32x8 acc[4][2] = {};
  gemm_pass(A, Bt, row0, col0, lA[0], lA[1], lB[0], lB[1], acc, tid);

  const int wm = (tid >> 5) >> 2, wn = (tid >> 5) & 3;
  const int mhalf = (lane >> 4) * 8, ncol = lane & 15;
#pragma unroll
  for (int i = 0; i < 4; ++i)
#pragma unroll
    for (int jj = 0; jj < 2; ++jj) {
      const int rbase = row0 + wm * 64 + i * 16 + mhalf;
      const int cbase = col0 + wn * 32 + jj * 16 + ncol;
#pragma unroll
      for (int r = 0; r < 8; ++r)
        out[(size_t)(rbase + r) * TDIM + cbase] = (__bf16)acc[i][jj][r];
    }
}

// --- pass C: pre_p = U_{2p}@R_{2p} + U_{2p+1}@R_{2p+1} + biases, f32 out ---
__global__ __launch_bounds__(NTHREADS) void klstm_gemm2(
    Ptrs8 bias, const __bf16* __restrict__ U, const __bf16* __restrict__ Tbf,
    float* __restrict__ pre) {
  __shared__ __bf16 lA[2][BM * LDS_STRIDE];
  __shared__ __bf16 lB[2][BN * LDS_STRIDE];

  const int tid  = threadIdx.x;
  const int lane = tid & 31;
  const int p    = blockIdx.z;   // 0=i 1=f 2=g 3=o
  const int row0 = blockIdx.y * BM;
  const int col0 = blockIdx.x * BN;

  f32x8 acc[4][2] = {};
  for (int phase = 0; phase < 2; ++phase) {
    const int g = 2 * p + phase;
    gemm_pass(U + (size_t)g * NM, Tbf + (size_t)(2 + g) * NM, row0, col0,
              lA[0], lA[1], lB[0], lB[1], acc, tid);
  }

  const float* bx = bias.p[2 * p];
  const float* bh = bias.p[2 * p + 1];
  float* out      = pre + (size_t)p * NM;
  const int wm = (tid >> 5) >> 2, wn = (tid >> 5) & 3;
  const int mhalf = (lane >> 4) * 8, ncol = lane & 15;
#pragma unroll
  for (int i = 0; i < 4; ++i)
#pragma unroll
    for (int jj = 0; jj < 2; ++jj) {
      const int rbase = row0 + wm * 64 + i * 16 + mhalf;
      const int cbase = col0 + wn * 32 + jj * 16 + ncol;
#pragma unroll
      for (int r = 0; r < 8; ++r) {
        const size_t idx = (size_t)(rbase + r) * TDIM + cbase;
        out[idx] = acc[i][jj][r] + bx[idx] + bh[idx];
      }
    }
}

// --- pass D: elementwise LSTM cell -----------------------------------------
__global__ __launch_bounds__(256) void klstm_cell(const float* __restrict__ pre,
                                                  const float* __restrict__ c,
                                                  float* __restrict__ out) {
  const size_t idx = (size_t)blockIdx.x * 256 + threadIdx.x;
  const float pi = pre[idx];
  const float pf = pre[idx + NM];
  const float pg = pre[idx + 2 * NM];
  const float po = pre[idx + 3 * NM];
  const float ig = 1.0f / (1.0f + expf(-pi));
  const float fg = 1.0f / (1.0f + expf(-pf));
  const float gg = tanhf(pg);
  const float og = 1.0f / (1.0f + expf(-po));
  const float cn = fg * c[idx] + ig * gg;
  const float hn = og * tanhf(cn);
  out[idx]      = hn;   // h_new
  out[idx + NM] = cn;   // c_new
}

extern "C" void kernel_launch(void* const* d_in, const int* in_sizes, int n_in,
                              void* d_out, int out_size, void* d_ws,
                              size_t ws_size, hipStream_t stream) {
  // inputs: 0=x 1=h 2=c, then per gate g in {ii,hi,if,hf,ig,hg,io,ho}:
  //   3+3g = L_g, 4+3g = R_g, 5+3g = b_g
  Ptrs8 Ls, biases;
  Ptrs10 Ts;
  Ts.p[0] = (const float*)d_in[0];  // x  -> Tbf slot 0
  Ts.p[1] = (const float*)d_in[1];  // h  -> Tbf slot 1
  const float* c = (const float*)d_in[2];
  for (int g = 0; g < 8; ++g) {
    Ls.p[g]     = (const float*)d_in[3 + 3 * g];
    Ts.p[2 + g] = (const float*)d_in[3 + 3 * g + 1];  // R_g -> Tbf slot 2+g
    biases.p[g] = (const float*)d_in[3 + 3 * g + 2];
  }

  // workspace: Lbf(8) | Tbf(10) | U(8) bf16, then pre(4) f32  = 68 MB
  __bf16* Lbf = (__bf16*)d_ws;
  __bf16* Tbf = Lbf + 8 * NM;
  __bf16* U   = Tbf + 10 * NM;
  float*  pre = (float*)(U + 8 * NM);

  dim3 blk(NTHREADS);
  klstm_conv<<<dim3(NM / (NTHREADS * 8), 1, 8), blk, 0, stream>>>(Ls, Lbf);
  klstm_convT<<<dim3(16, 16, 10), blk, 0, stream>>>(Ts, Tbf);
  klstm_gemm1<<<dim3(TDIM / BN, TDIM / BM, 8), blk, 0, stream>>>(Lbf, Tbf, U);
  klstm_gemm2<<<dim3(TDIM / BN, TDIM / BM, 4), blk, 0, stream>>>(biases, U, Tbf,
                                                                 pre);
  klstm_cell<<<(unsigned)(NM / 256), 256, 0, stream>>>(pre, c, (float*)d_out);
}
